// MultiHeadedAttention_15049565405595
// MI455X (gfx1250) — matrix operand order repaired and checked by reading, not compiled, not run on hardware
//
#include <hip/hip_runtime.h>

typedef __attribute__((ext_vector_type(16))) __bf16 v16bf;
typedef __attribute__((ext_vector_type(8)))  __bf16 v8bf;
typedef __attribute__((ext_vector_type(4)))  __bf16 v4bf;
typedef __attribute__((ext_vector_type(8)))  float  v8f;
typedef __attribute__((ext_vector_type(4)))  float  v4f;

// Type matching the async-to-LDS builtin's parameter: pointer to <4 x i32>
typedef int v4i_vs __attribute__((vector_size(16)));

#define HD     16
#define DMODEL 1024
#define DK     64
#define BATCH  4
#define SEQ    2048
#define MROWS  (BATCH * SEQ)   // 8192

// ---------------------------------------------------------------------------
// CDNA5 async global->LDS staging (ASYNCcnt path), with safe fallback.
// Builtin signature (probe-confirmed via round-2 diagnostic):
//   void __builtin_amdgcn_global_load_async_to_lds_b128(
//        v4i_vs addrspace(1)* gsrc, v4i_vs addrspace(3)* ldst,
//        imm int offset, imm int cpol)
// ---------------------------------------------------------------------------
#if __has_builtin(__builtin_amdgcn_global_load_async_to_lds_b128) && \
    __has_builtin(__builtin_amdgcn_s_wait_asynccnt)
#define HAVE_ASYNC_LDS 1
#endif

__device__ __forceinline__ void stage16(const __bf16* g, __bf16* l)
{
#ifdef HAVE_ASYNC_LDS
    __builtin_amdgcn_global_load_async_to_lds_b128(
        (__attribute__((address_space(1))) v4i_vs*)(void*)g,
        (__attribute__((address_space(3))) v4i_vs*)(void*)l,
        0, 0);
#else
    *(v8bf*)l = *(const v8bf*)g;
#endif
}

__device__ __forceinline__ void stage_wait()
{
#ifdef HAVE_ASYNC_LDS
    __builtin_amdgcn_s_wait_asynccnt(0);
#endif
}

// ---------------------------------------------------------------------------
// fp32 -> bf16 conversion (bandwidth-bound pre-pass, activations)
// ---------------------------------------------------------------------------
__global__ __launch_bounds__(256)
void cvt_f32_to_bf16(const float* __restrict__ in, __bf16* __restrict__ out, int n4)
{
    int i = blockIdx.x * blockDim.x + threadIdx.x;
    if (i < n4) {
        v4f v = ((const v4f*)in)[i];
        v4bf o;
        o[0] = (__bf16)v.x; o[1] = (__bf16)v.y;
        o[2] = (__bf16)v.z; o[3] = (__bf16)v.w;
        ((v4bf*)out)[i] = o;
    }
}

// ---------------------------------------------------------------------------
// fp32 W[k][n] -> bf16 Wt[n][k] one-time transpose (LDS tile transpose).
// Hoists the B-operand transpose out of every GEMM k-step.
// ---------------------------------------------------------------------------
__global__ __launch_bounds__(256)
void cvt_w_transpose(const float* __restrict__ W, __bf16* __restrict__ Wt)
{
    __shared__ __bf16 tile[64 * 68];
    const int t  = threadIdx.x;
    const int k0 = blockIdx.y * 64;
    const int n0 = blockIdx.x * 64;
    const int r  = t >> 4;           // 0..15
    const int c4 = (t & 15) * 4;     // 0..60

    #pragma unroll
    for (int p = 0; p < 4; ++p) {
        int k = r + p * 16;
        v4f v = *(const v4f*)(W + (size_t)(k0 + k) * DMODEL + n0 + c4);
        tile[k * 68 + c4 + 0] = (__bf16)v.x;
        tile[k * 68 + c4 + 1] = (__bf16)v.y;
        tile[k * 68 + c4 + 2] = (__bf16)v.z;
        tile[k * 68 + c4 + 3] = (__bf16)v.w;
    }
    __syncthreads();
    #pragma unroll
    for (int p = 0; p < 4; ++p) {
        int n = r + p * 16;
        v4bf o;
        #pragma unroll
        for (int i = 0; i < 4; ++i) o[i] = tile[(c4 + i) * 68 + n];
        *(v4bf*)(Wt + (size_t)(n0 + n) * DMODEL + k0 + c4) = o;
    }
}

// ---------------------------------------------------------------------------
// bf16 WMMA GEMM:  C[8192 x 1024] = A[8192 x 1024] * Wt^T + bias
//   A  : [M][K] bf16 row-major
//   Bt : [N][K] bf16 (pre-transposed weights) -> contiguous b128 staging
//   mode 0: store bf16 to [b,h,s,d]   (Q with scale=0.125, K with scale=1)
//   mode 1: store bf16 to [b,h,d,s]   (V transposed for attention B-operand)
//   mode 2: store fp32 to [r,f]       (final output projection)
// Block tile 128x128, BK=32, 256 threads = 8 waves (2x4), wave tile 64x32.
// LDS row stride 40 bf16 (80 B): 16B-aligned, conflict-free fragment reads.
// ---------------------------------------------------------------------------
__global__ __launch_bounds__(256)
void gemm_bf16(const __bf16* __restrict__ A, const __bf16* __restrict__ Bt,
               const float* __restrict__ bias, void* __restrict__ outp,
               int mode, float scale)
{
    const int N = DMODEL, K = DMODEL;
    __shared__ __bf16 Alds[128 * 40];   // [m][k]
    __shared__ __bf16 Blds[128 * 40];   // [n][k]

    const int t       = threadIdx.x;
    const int rowBase = blockIdx.y * 128;
    const int colBase = blockIdx.x * 128;
    const int w  = t >> 5,  lane = t & 31;
    const int wm = w >> 2,  wn   = w & 3;
    const int hi = lane >> 4, ln = lane & 15;

    v8f acc[4][2];
    #pragma unroll
    for (int mi = 0; mi < 4; ++mi)
        #pragma unroll
        for (int ni = 0; ni < 2; ++ni)
            #pragma unroll
            for (int v = 0; v < 8; ++v) acc[mi][ni][v] = 0.0f;

    const int am = t >> 2;          // 0..63 (+64 on 2nd pass)
    const int ak = (t & 3) * 8;     // {0,8,16,24}

    for (int k0 = 0; k0 < K; k0 += 32) {
        // ---- stage A tile [128 x 32] and B tile [128 n x 32 k], async to LDS
        #pragma unroll
        for (int p = 0; p < 2; ++p) {
            int m = am + p * 64;
            stage16(A  + (size_t)(rowBase + m) * K + k0 + ak, &Alds[m * 40 + ak]);
            stage16(Bt + (size_t)(colBase + m) * K + k0 + ak, &Blds[m * 40 + ak]);
        }
        if (k0 + 32 < K) {
            __builtin_prefetch(A  + (size_t)(rowBase + am) * K + k0 + 32 + ak, 0, 1);
            __builtin_prefetch(Bt + (size_t)(colBase + am) * K + k0 + 32 + ak, 0, 1);
        }
        stage_wait();
        __syncthreads();

        // ---- B fragments (32x16: lane = col n, halves interleave k)
        v16bf bfrag[2];
        #pragma unroll
        for (int ni = 0; ni < 2; ++ni) {
            int n = wn * 32 + ni * 16 + ln;
            v8bf blo = *(const v8bf*)(&Blds[n * 40 + hi * 8]);
            v8bf bh2 = *(const v8bf*)(&Blds[n * 40 + 16 + hi * 8]);
            #pragma unroll
            for (int i = 0; i < 8; ++i) { bfrag[ni][i] = blo[i]; bfrag[ni][i + 8] = bh2[i]; }
        }
        // ---- A fragments (16x32: lane = row m, halves interleave k) + WMMA
        #pragma unroll
        for (int mi = 0; mi < 4; ++mi) {
            int m = wm * 64 + mi * 16 + ln;
            v8bf alo = *(const v8bf*)(&Alds[m * 40 + hi * 8]);
            v8bf ahi = *(const v8bf*)(&Alds[m * 40 + 16 + hi * 8]);
            v16bf afrag;
            #pragma unroll
            for (int i = 0; i < 8; ++i) { afrag[i] = alo[i]; afrag[i + 8] = ahi[i]; }
            #pragma unroll
            for (int ni = 0; ni < 2; ++ni)
                acc[mi][ni] = __builtin_amdgcn_wmma_f32_16x16x32_bf16(
                    false, afrag, false, bfrag[ni], (short)0, acc[mi][ni], false, false);
        }
        __syncthreads();
    }

    // ---- epilogue: C layout is (row m = v + 8*hi, col n = ln)
    #pragma unroll
    for (int mi = 0; mi < 4; ++mi) {
        #pragma unroll
        for (int ni = 0; ni < 2; ++ni) {
            int f = colBase + wn * 32 + ni * 16 + ln;
            float bsv = bias[f];
            #pragma unroll
            for (int v = 0; v < 8; ++v) {
                int r = rowBase + wm * 64 + mi * 16 + v + 8 * hi;
                float val = (acc[mi][ni][v] + bsv) * scale;
                if (mode == 2) {
                    ((float*)outp)[(size_t)r * N + f] = val;
                } else {
                    int b = r >> 11, s = r & (SEQ - 1);
                    int h = f >> 6,  d = f & (DK - 1);
                    size_t idx = (mode == 0)
                        ? (((size_t)(b * HD + h)) * SEQ + s) * DK + d    // [b,h,s,d]
                        : (((size_t)(b * HD + h)) * DK + d) * SEQ + s;   // [b,h,d,s]
                    ((__bf16*)outp)[idx] = (__bf16)val;
                }
            }
        }
    }
}

// ---------------------------------------------------------------------------
// Flash attention: one block = 128 query rows of one (b,h); 8 waves x 16 rows.
// Q pre-scaled by 1/sqrt(dk). K in [b,h,s,d], V in [b,h,d,s].
// Online softmax via 16-lane shfl_xor butterflies (a C-layout row lives in
// lanes {hi*16 .. hi*16+15}); P re-laid out C->A through per-wave LDS.
// ---------------------------------------------------------------------------
__global__ __launch_bounds__(256)
void attn_flash(const __bf16* __restrict__ Qp, const __bf16* __restrict__ Kp,
                const __bf16* __restrict__ Vt, __bf16* __restrict__ outc)
{
    __shared__ __bf16 Klds[64 * 72];        // [kv][d]  stride 72 (144 B)
    __shared__ __bf16 Vlds[64 * 72];        // [d][kv]
    __shared__ __bf16 Plds[8 * 16 * 72];    // per wave: [m][kv]

    const int t  = threadIdx.x;
    const int w  = t >> 5, lane = t & 31;
    const int hi = lane >> 4, ln = lane & 15;
    const int bh = blockIdx.y;              // 0..63
    const int q0 = blockIdx.x * 128;

    const __bf16* Qb = Qp + (size_t)bh * SEQ * DK;
    const __bf16* Kb = Kp + (size_t)bh * SEQ * DK;
    const __bf16* Vb = Vt + (size_t)bh * DK * SEQ;

    // Q fragments (A layout), resident for the whole kernel
    v16bf qA[2];
    #pragma unroll
    for (int c = 0; c < 2; ++c) {
        const __bf16* qr = Qb + (size_t)(q0 + w * 16 + ln) * DK + c * 32;
        v8bf lo  = *(const v8bf*)(qr + hi * 8);
        v8bf hi8 = *(const v8bf*)(qr + 16 + hi * 8);
        #pragma unroll
        for (int i = 0; i < 8; ++i) { qA[c][i] = lo[i]; qA[c][i + 8] = hi8[i]; }
    }

    v8f Oacc[4];
    float mrow[8], lrow[8];
    #pragma unroll
    for (int tt = 0; tt < 4; ++tt)
        #pragma unroll
        for (int v = 0; v < 8; ++v) Oacc[tt][v] = 0.0f;
    #pragma unroll
    for (int v = 0; v < 8; ++v) { mrow[v] = -1e30f; lrow[v] = 0.0f; }

    const int krow = t >> 3;          // 0..31 (+32 on 2nd pass)
    const int kcol = (t & 7) * 8;     // 0..56

    for (int kv0 = 0; kv0 < SEQ; kv0 += 64) {
        __syncthreads();   // prior-iteration LDS readers done
        #pragma unroll
        for (int p = 0; p < 2; ++p) {
            int rr = krow + p * 32;
            stage16(Kb + (size_t)(kv0 + rr) * DK + kcol, &Klds[rr * 72 + kcol]);
            stage16(Vb + (size_t)rr * SEQ + kv0 + kcol,  &Vlds[rr * 72 + kcol]);
        }
        stage_wait();
        __syncthreads();

        // ---- S = Q * K^T  (16 x 64 per wave, 8 WMMAs)
        v8f S[4];
        #pragma unroll
        for (int tt = 0; tt < 4; ++tt)
            #pragma unroll
            for (int v = 0; v < 8; ++v) S[tt][v] = 0.0f;
        #pragma unroll
        for (int c = 0; c < 2; ++c) {
            #pragma unroll
            for (int tt = 0; tt < 4; ++tt) {
                int n = tt * 16 + ln;     // kv column
                v8bf blo = *(const v8bf*)(&Klds[n * 72 + c * 32 + hi * 8]);
                v8bf bh2 = *(const v8bf*)(&Klds[n * 72 + c * 32 + 16 + hi * 8]);
                v16bf bf;
                #pragma unroll
                for (int i = 0; i < 8; ++i) { bf[i] = blo[i]; bf[i + 8] = bh2[i]; }
                S[tt] = __builtin_amdgcn_wmma_f32_16x16x32_bf16(
                    false, qA[c], false, bf, (short)0, S[tt], false, false);
            }
        }

        // ---- online softmax (per C-layout row m = v + 8*hi)
        float mnew[8], pscale[8], tsum[8];
        #pragma unroll
        for (int v = 0; v < 8; ++v) {
            float mx = fmaxf(fmaxf(S[0][v], S[1][v]), fmaxf(S[2][v], S[3][v]));
            #pragma unroll
            for (int d = 1; d <= 8; d <<= 1)
                mx = fmaxf(mx, __shfl_xor(mx, d, 32));
            mnew[v]   = fmaxf(mrow[v], mx);
            pscale[v] = __expf(mrow[v] - mnew[v]);
            tsum[v]   = 0.0f;
        }
        #pragma unroll
        for (int tt = 0; tt < 4; ++tt) {
            #pragma unroll
            for (int v = 0; v < 8; ++v) {
                float p = __expf(S[tt][v] - mnew[v]);
                tsum[v] += p;
                Plds[w * 1152 + (v + 8 * hi) * 72 + tt * 16 + ln] = (__bf16)p;
            }
        }
        #pragma unroll
        for (int v = 0; v < 8; ++v) {
            float s = tsum[v];
            #pragma unroll
            for (int d = 1; d <= 8; d <<= 1)
                s += __shfl_xor(s, d, 32);
            lrow[v] = lrow[v] * pscale[v] + s;
            mrow[v] = mnew[v];
        }
        #pragma unroll
        for (int tt = 0; tt < 4; ++tt)
            #pragma unroll
            for (int v = 0; v < 8; ++v) Oacc[tt][v] *= pscale[v];

        __syncthreads();   // P visible to A-layout reads

        // ---- O += P * V  (8 WMMAs; B operand contiguous thanks to Vt layout)
        #pragma unroll
        for (int c = 0; c < 2; ++c) {
            const __bf16* pr = &Plds[w * 1152 + ln * 72 + c * 32];
            v8bf plo = *(const v8bf*)(pr + hi * 8);
            v8bf phi = *(const v8bf*)(pr + 16 + hi * 8);
            v16bf pA;
            #pragma unroll
            for (int i = 0; i < 8; ++i) { pA[i] = plo[i]; pA[i + 8] = phi[i]; }
            #pragma unroll
            for (int tt = 0; tt < 4; ++tt) {
                int n = tt * 16 + ln;     // d column
                v8bf blo = *(const v8bf*)(&Vlds[n * 72 + c * 32 + hi * 8]);
                v8bf bh2 = *(const v8bf*)(&Vlds[n * 72 + c * 32 + 16 + hi * 8]);
                v16bf bf;
                #pragma unroll
                for (int i = 0; i < 8; ++i) { bf[i] = blo[i]; bf[i + 8] = bh2[i]; }
                Oacc[tt] = __builtin_amdgcn_wmma_f32_16x16x32_bf16(
                    false, pA, false, bf, (short)0, Oacc[tt], false, false);
            }
        }
    }

    // ---- normalize and store merged-head layout [B, S, H*DK] as bf16
    const int b = bh >> 4, h = bh & 15;
    #pragma unroll
    for (int v = 0; v < 8; ++v) {
        float inv = 1.0f / lrow[v];
        int s = q0 + w * 16 + v + 8 * hi;
        size_t rowoff = ((size_t)(b * SEQ + s)) * DMODEL + h * DK;
        #pragma unroll
        for (int tt = 0; tt < 4; ++tt)
            outc[rowoff + tt * 16 + ln] = (__bf16)(Oacc[tt][v] * inv);
    }
}

// ---------------------------------------------------------------------------
// Host-side orchestration
// ---------------------------------------------------------------------------
extern "C" void kernel_launch(void* const* d_in, const int* in_sizes, int n_in,
                              void* d_out, int out_size, void* d_ws, size_t ws_size,
                              hipStream_t stream)
{
    (void)in_sizes; (void)n_in; (void)out_size; (void)ws_size;

    const float* q_in = (const float*)d_in[0];
    const float* k_in = (const float*)d_in[1];
    const float* v_in = (const float*)d_in[2];
    const float* Wq = (const float*)d_in[3];  const float* bq = (const float*)d_in[4];
    const float* Wk = (const float*)d_in[5];  const float* bk = (const float*)d_in[6];
    const float* Wv = (const float*)d_in[7];  const float* bv = (const float*)d_in[8];
    const float* Wo = (const float*)d_in[9];  const float* bo = (const float*)d_in[10];

    const size_t NX = (size_t)MROWS * DMODEL;    // 8,388,608 elements
    const size_t NW = (size_t)DMODEL * DMODEL;   // 1,048,576 elements

    char* ws = (char*)d_ws;
    size_t off = 0;
    __bf16* xq  = (__bf16*)(ws + off); off += NX * 2;
    __bf16* xk  = (__bf16*)(ws + off); off += NX * 2;
    __bf16* xv  = (__bf16*)(ws + off); off += NX * 2;
    __bf16* wqT = (__bf16*)(ws + off); off += NW * 2;   // [n][k]
    __bf16* wkT = (__bf16*)(ws + off); off += NW * 2;
    __bf16* wvT = (__bf16*)(ws + off); off += NW * 2;
    __bf16* woT = (__bf16*)(ws + off); off += NW * 2;
    __bf16* Qp  = (__bf16*)(ws + off); off += NX * 2;   // [b,h,s,d] pre-scaled
    __bf16* Kp  = (__bf16*)(ws + off); off += NX * 2;   // [b,h,s,d]
    __bf16* Vtp = (__bf16*)(ws + off); off += NX * 2;   // [b,h,d,s]
    __bf16* cc  = (__bf16*)(ws + off); off += NX * 2;   // merged heads [B,S,D]

    // 1) fp32 -> bf16 conversions; weights transposed once to [n][k]
    int nx4 = (int)(NX / 4);
    cvt_f32_to_bf16<<<nx4 / 256, 256, 0, stream>>>(q_in, xq, nx4);
    cvt_f32_to_bf16<<<nx4 / 256, 256, 0, stream>>>(k_in, xk, nx4);
    cvt_f32_to_bf16<<<nx4 / 256, 256, 0, stream>>>(v_in, xv, nx4);
    dim3 tgrid(DMODEL / 64, DMODEL / 64);    // (16,16)
    cvt_w_transpose<<<tgrid, 256, 0, stream>>>(Wq, wqT);
    cvt_w_transpose<<<tgrid, 256, 0, stream>>>(Wk, wkT);
    cvt_w_transpose<<<tgrid, 256, 0, stream>>>(Wv, wvT);
    cvt_w_transpose<<<tgrid, 256, 0, stream>>>(Wo, woT);

    // 2) projections (Q scaled by 1/sqrt(dk) = 0.125; V stored transposed)
    dim3 ggrid(DMODEL / 128, MROWS / 128);   // (8, 64)
    gemm_bf16<<<ggrid, 256, 0, stream>>>(xq, wqT, bq, (void*)Qp,  0, 0.125f);
    gemm_bf16<<<ggrid, 256, 0, stream>>>(xk, wkT, bk, (void*)Kp,  0, 1.0f);
    gemm_bf16<<<ggrid, 256, 0, stream>>>(xv, wvT, bv, (void*)Vtp, 1, 1.0f);

    // 3) flash attention: grid (q-tiles, b*h) = (16, 64)
    dim3 agrid(SEQ / 128, BATCH * HD);
    attn_flash<<<agrid, 256, 0, stream>>>(Qp, Kp, Vtp, cc);

    // 4) output projection, fp32 result + bias
    gemm_bf16<<<ggrid, 256, 0, stream>>>(cc, woT, bo, d_out, 2, 1.0f);
}
